// RNN_80848464380442
// MI455X (gfx1250) — matrix operand order repaired and checked by reading, not compiled, not run on hardware
//
#include <hip/hip_runtime.h>
#include <hip/hip_bf16.h>

typedef __attribute__((ext_vector_type(2)))  float  v2f;
typedef __attribute__((ext_vector_type(8)))  float  v8f;
typedef __attribute__((ext_vector_type(16))) __bf16 v16bf;
typedef __attribute__((ext_vector_type(8)))  __bf16 v8bf;

#define B_ 64
#define S_ 2048
#define E_ 128
#define H_ 256
#define C_ 2
#define APAD 132      // kernel A LDS row stride (floats): conflict-free, 8B-aligned
#define RH  264       // kernel B h row stride (bf16): 16B-aligned rows, bank-spread

// ---------------------------------------------------------------------------
// Kernel A: xw[s*B + b][h] = sum_e emb_eff[x[b,s]][e] * W[e][h]   (fp32 WMMA)
// One block = one 16-row M-tile of (s,b); 8 waves x 2 N-tiles cover H=256.
// ---------------------------------------------------------------------------
__global__ __launch_bounds__(256) void xw_gemm_kernel(
    const int* __restrict__ x, const float* __restrict__ emb,
    const float* __restrict__ W, float* __restrict__ xw)
{
  __shared__ int   toks[16];
  __shared__ float tileA[16 * APAD];

  const int tid  = threadIdx.x;
  const int lane = tid & 31;
  const int wave = tid >> 5;
  const int r0   = blockIdx.x * 16;          // global row = s*64 + b

  if (tid < 16) {
    int r = r0 + tid;
    int s = r >> 6;                           // r / 64
    int b = r & 63;
    toks[tid] = x[b * S_ + s];
  }
  __syncthreads();

  for (int idx = tid; idx < 16 * E_; idx += 256) {
    int m = idx >> 7;
    int k = idx & 127;
    int t = toks[m];
    tileA[m * APAD + k] = (t == 0) ? 0.0f : emb[(long)t * E_ + k];   // padding_idx=0
  }
  __syncthreads();

  const int mrow  = lane & 15;
  const int khalf = (lane >> 4) * 2;          // f32 A/B: lanes 16-31 hold K=2,3
  const int hih   = lane >> 4;

  for (int nt = 0; nt < 2; ++nt) {
    const int n0 = (wave * 2 + nt) * 16;
    v8f c = {};
    #pragma unroll
    for (int kk = 0; kk < E_ / 4; ++kk) {
      const int kb = kk * 4 + khalf;
      v2f a = *(const v2f*)&tileA[mrow * APAD + kb];     // ds_load_b64
      v2f bm;
      bm.x = W[(kb + 0) * H_ + n0 + mrow];
      bm.y = W[(kb + 1) * H_ + n0 + mrow];
      c = __builtin_amdgcn_wmma_f32_16x16x4_f32(false, a, false, bm,
                                                (short)0, c, false, false);
    }
    #pragma unroll
    for (int j = 0; j < 8; ++j) {
      int m = j + 8 * hih;                    // C layout: VGPR j -> M=j / j+8
      xw[(long)(r0 + m) * H_ + n0 + mrow] = c[j];
    }
  }
}

// ---------------------------------------------------------------------------
// Kernel B: the serial scan, bf16 WMMA (f32 accumulate).
// 4 blocks (16 batch rows each), 16 waves; wave w owns H columns [16w,16w+16).
// U held in VGPRs (64/lane, loop-invariant); h double-buffered in LDS as bf16
// in WMMA-A-friendly layout -> 2 ds_load_b128 per WMMA, 8 WMMAs per step.
// ---------------------------------------------------------------------------
__global__ __launch_bounds__(512) void rnn_scan_kernel(
    const float* __restrict__ xw, const float* __restrict__ U,
    const float* __restrict__ bU, const float* __restrict__ Vm,
    const float* __restrict__ bV, float* __restrict__ out)
{
  __shared__ __bf16 hbuf[2 * 16 * RH];        // double-buffered h, ~17 KB

  const int tid  = threadIdx.x;
  const int lane = tid & 31;
  const int wave = tid >> 5;                  // 0..15
  const int bt   = blockIdx.x;                // 0..3 batch tile
  const int n0   = wave * 16;
  const int mrow = lane & 15;
  const int hih  = lane >> 4;

  // Preload this wave's 16-column slice of U as the bf16 B operand, in VGPRs.
  // B layout (32x16 bf16): lane n = l&15; element e of v16bf ->
  // K = 32*kk + 8*hih + e (e<8), K = 32*kk + 16 + 8*hih + (e-8) (e>=8).
  v16bf bop[8];
  #pragma unroll
  for (int kk = 0; kk < 8; ++kk) {
    #pragma unroll
    for (int e = 0; e < 16; ++e) {
      bop[kk][e] = (__bf16)U[(32 * kk + (e < 8 ? 8 * hih + e : 16 + 8 * hih + (e - 8))) * H_ + n0 + mrow];
    }
  }

  // h0 = 0 in buffer 0
  for (int idx = tid; idx < 16 * RH; idx += 512) {
    hbuf[idx] = (__bf16)0.0f;
  }
  const float bias = bU[n0 + mrow];
  __syncthreads();

  for (int s = 0; s < S_; ++s) {
    const __bf16* hR = hbuf + ((s) & 1) * (16 * RH);
    __bf16*       hW = hbuf + ((s + 1) & 1) * (16 * RH);

    // issue this step's fp32 xw loads up front (L2-resident; hide under WMMA)
    float xwv[8];
    long base = ((long)s * B_ + bt * 16) * H_ + n0 + mrow;
    #pragma unroll
    for (int j = 0; j < 8; ++j) {
      xwv[j] = xw[base + (long)(j + 8 * hih) * H_];
    }

    v8f c = {};
    #pragma unroll
    for (int kk = 0; kk < 8; ++kk) {
      // A layout (16x32 bf16): lane row m = l&15; two contiguous 8xbf16
      // chunks: K = 32kk + 8*hih + {0..7} and K = 32kk + 16 + 8*hih + {0..7}
      const int kb = 32 * kk + 8 * hih;
      v8bf a0 = *(const v8bf*)&hR[mrow * RH + kb];        // ds_load_b128
      v8bf a1 = *(const v8bf*)&hR[mrow * RH + kb + 16];   // ds_load_b128
      v16bf a;
      #pragma unroll
      for (int e = 0; e < 8; ++e) { a[e] = a0[e]; a[e + 8] = a1[e]; }
      c = __builtin_amdgcn_wmma_f32_16x16x32_bf16(false, a, false, bop[kk],
                                                  (short)0, c, false, false);
    }

    #pragma unroll
    for (int j = 0; j < 8; ++j) {
      int m = j + 8 * hih;
      hW[m * RH + n0 + mrow] = (__bf16)tanhf(c[j] + xwv[j] + bias);
    }
    __syncthreads();
  }

  const __bf16* hF = hbuf;                    // final h lands in buffer 0

  // h -> d_out[128 + ...]
  for (int idx = tid; idx < 16 * H_; idx += 512) {
    int m = idx >> 8;
    int n = idx & (H_ - 1);
    out[128 + (long)(bt * 16 + m) * H_ + n] = (float)hF[m * RH + n];
  }
  // logits = h @ V + bV   (16x2 per block; one wave handles it)
  if (tid < 32) {
    int row = tid & 15;
    int col = tid >> 4;
    float sum = bV[col];
    for (int k = 0; k < H_; ++k) {
      sum += (float)hF[row * RH + k] * Vm[k * C_ + col];
    }
    out[(bt * 16 + row) * C_ + col] = sum;
  }
}

// ---------------------------------------------------------------------------
extern "C" void kernel_launch(void* const* d_in, const int* in_sizes, int n_in,
                              void* d_out, int out_size, void* d_ws, size_t ws_size,
                              hipStream_t stream) {
  const int*   x   = (const int*)d_in[0];
  const float* emb = (const float*)d_in[1];
  const float* W   = (const float*)d_in[2];
  const float* U   = (const float*)d_in[3];
  const float* bU  = (const float*)d_in[4];
  const float* Vm  = (const float*)d_in[5];
  const float* bV  = (const float*)d_in[6];
  float* out = (float*)d_out;
  float* xw  = (float*)d_ws;                  // [S*B*H] f32 = 134 MB scratch (fits L2)

  xw_gemm_kernel<<<dim3((S_ * B_) / 16), 256, 0, stream>>>(x, emb, W, xw);
  rnn_scan_kernel<<<dim3(4), 512, 0, stream>>>(xw, U, bU, Vm, bV, out);
}
